// AutoEncoderLSTM_43490838839396
// MI455X (gfx1250) — matrix operand order repaired and checked by reading, not compiled, not run on hardware
//
#include <hip/hip_runtime.h>

// AutoEncoder LSTM for MI455X (gfx1250), wave32 + WMMA + native V_TANH.
// B=4096, T=2048, H=64, IN=1, BOT=3.
//
// Each workgroup (128 threads = 4 waves) owns ROWS=16 batch rows persistently:
//   wave g computes gate g (64 of the 256 gate outputs) for those 16 rows via
//   v_wmma_f32_16x16x32_f16:  g[16x256] = h16[16x64] @ Whh^T  (2 k-steps)
//                                       + [x,1] @ [Wih;b]     (augmented 3rd WMMA)
// h (f16) and c (f32) live in LDS across all 4096 recurrent steps.

#define BATCH   4096
#define SEQT    2048
#define HID     64
#define GATES   256
#define ROWS    16
#define NTHREADS 128

typedef __attribute__((ext_vector_type(16))) _Float16 v16h;
typedef __attribute__((ext_vector_type(8)))  _Float16 v8h;
typedef __attribute__((ext_vector_type(8)))  float    v8f;

__device__ __forceinline__ v8f wmma32(v16h a, v16h b, v8f c) {
  // (neg_a, A, neg_b, B, c_mod, C, reuse_a, reuse_b)
  return __builtin_amdgcn_wmma_f32_16x16x32_f16(false, a, false, b, (short)0, c,
                                                false, false);
}

// gfx1250 has a native transcendental V_TANH_F32 (ISA ch.8 TRANS list).
__device__ __forceinline__ float fast_tanh(float x) {
#if __has_builtin(__builtin_amdgcn_tanhf)
  return __builtin_amdgcn_tanhf(x);
#else
  float r;
  asm volatile("v_tanh_f32 %0, %1" : "=v"(r) : "v"(x));
  return r;
#endif
}

// sigmoid(x) = 0.5 * tanh(x/2) + 0.5  -> 1 TRANS + 2 VALU, branch-free.
__device__ __forceinline__ float fast_sigmoid(float x) {
  return __builtin_fmaf(0.5f, fast_tanh(0.5f * x), 0.5f);
}

// B fragment (K=32 x N=16 tile) from f32 weight matrix W[GATES][HID], column n
// of B == row n of W.  Layout (matches A table order): lanes 0-15 hold
// K=0..7,16..23; lanes 16-31 hold K=8..15,24..31.
__device__ __forceinline__ v16h load_bfrag(const float* __restrict__ W, int n,
                                           int kbase) {
  const int lane = threadIdx.x & 31;
  const int kh = (lane >= 16) ? 8 : 0;
  const float* p = W + n * HID + kbase + kh;
  v16h r;
#pragma unroll
  for (int j = 0; j < 16; ++j) r[j] = (_Float16)p[j < 8 ? j : j + 8];
  return r;
}

// Augmented-K B fragment: row K=0 -> Wih[n], row K=1 -> bias[n], rest zero.
__device__ __forceinline__ v16h make_bext(const float* __restrict__ Wih,
                                          const float* __restrict__ bias, int n) {
  const int lane = threadIdx.x & 31;
  const bool lo = lane < 16;
  v16h r;
#pragma unroll
  for (int j = 0; j < 16; ++j) r[j] = (_Float16)0.0f;
  r[0] = lo ? (_Float16)Wih[n]  : (_Float16)0.0f;
  r[1] = lo ? (_Float16)bias[n] : (_Float16)0.0f;
  return r;
}

// One LSTM cell step for this workgroup's 16 rows.
__device__ __forceinline__ void lstm_step(const v16h (&Bw)[4][2],
                                          const v16h (&Bx)[4], float xval,
                                          bool masked, int t,
                                          _Float16* hbuf, float* cbuf,
                                          float* gbuf, const int* lenbuf) {
  const int tid = threadIdx.x;
  const int lane = tid & 31;
  const int gate = tid >> 5;           // wave id == gate id (0..3)
  const int arow = lane & 15;          // A-matrix M row
  const int koff = (lane >> 4) << 3;   // 0 or 8 (ISA 16-bit A layout)

  // A fragments from LDS h16 (two ds_load_b128 per fragment).
  const v8h* hp = (const v8h*)(hbuf + arow * HID + koff);
  v8h l0 = hp[0], h0 = hp[2];          // k-step 0: K koff..+7, koff+16..+23
  v8h l1 = hp[4], h1 = hp[6];          // k-step 1 (+32 halves)
  v16h a0 = __builtin_shufflevector(l0, h0, 0, 1, 2, 3, 4, 5, 6, 7, 8, 9, 10,
                                    11, 12, 13, 14, 15);
  v16h a1 = __builtin_shufflevector(l1, h1, 0, 1, 2, 3, 4, 5, 6, 7, 8, 9, 10,
                                    11, 12, 13, 14, 15);
  // Augmented A: per row [x, 1, 0...]; lanes>=16 (K 8..15,24..31) are zero.
  const bool alo = lane < 16;
  v16h ax;
#pragma unroll
  for (int j = 0; j < 16; ++j) ax[j] = (_Float16)0.0f;
  ax[0] = alo ? (_Float16)xval : (_Float16)0.0f;
  ax[1] = alo ? (_Float16)1.0f : (_Float16)0.0f;

  const int rowoff = (lane >= 16) ? 8 : 0;
#pragma unroll
  for (int nt = 0; nt < 4; ++nt) {
    v8f acc;
#pragma unroll
    for (int i = 0; i < 8; ++i) acc[i] = 0.0f;
    acc = wmma32(a0, Bw[nt][0], acc);
    acc = wmma32(a1, Bw[nt][1], acc);
    acc = wmma32(ax, Bx[nt], acc);
    // C layout: vgpr r, lane -> (M = r + rowoff, N = lane&15).
    const int col = gate * 64 + nt * 16 + (lane & 15);
    float* gp = gbuf + col * ROWS + rowoff;
    float4 p0 = make_float4(acc[0], acc[1], acc[2], acc[3]);
    float4 p1 = make_float4(acc[4], acc[5], acc[6], acc[7]);
    ((float4*)gp)[0] = p0;
    ((float4*)gp)[1] = p1;
  }
  __syncthreads();

  // Cell update: thread (row = tid&15) handles 8 consecutive k.
  // 5 native v_tanh per element, branch-free.
  const int urow = tid & 15;
  const int ukb = (tid >> 4) << 3;
#pragma unroll
  for (int j = 0; j < 8; ++j) {
    const int k = ukb + j;
    float gi = gbuf[(0 * HID + k) * ROWS + urow];
    float gf = gbuf[(1 * HID + k) * ROWS + urow];
    float gg = gbuf[(2 * HID + k) * ROWS + urow];
    float go = gbuf[(3 * HID + k) * ROWS + urow];
    float cold = cbuf[k * ROWS + urow];
    float cn = fast_sigmoid(gf) * cold + fast_sigmoid(gi) * fast_tanh(gg);
    float hn = fast_sigmoid(go) * fast_tanh(cn);
    if (masked) {
      bool act = t < lenbuf[urow];
      float hold = (float)hbuf[urow * HID + k];
      cn = act ? cn : cold;
      hn = act ? hn : hold;
    }
    cbuf[k * ROWS + urow] = cn;
    hbuf[urow * HID + k] = (_Float16)hn;
  }
  __syncthreads();
}

__global__ __launch_bounds__(NTHREADS) void lstm_ae_kernel(
    const float* __restrict__ pad, const int* __restrict__ seqlen,
    const float* __restrict__ enc_Wih, const float* __restrict__ enc_Whh,
    const float* __restrict__ enc_b, const float* __restrict__ enc_linW,
    const float* __restrict__ enc_linb, const float* __restrict__ dec_linW,
    const float* __restrict__ dec_linb, const float* __restrict__ dec_Wih,
    const float* __restrict__ dec_Whh, const float* __restrict__ dec_b,
    const float* __restrict__ outW, const float* __restrict__ outb,
    float* __restrict__ out_pad, float* __restrict__ out_y,
    float* __restrict__ out_hz, float* __restrict__ ws_sse) {
  __shared__ alignas(16) _Float16 sh_h[ROWS * HID];   // h state, [row][k], f16
  __shared__ alignas(16) float sh_c[HID * ROWS];      // c state, [k][row]
  __shared__ alignas(16) float sh_g[GATES * ROWS];    // gates,  [col][row]
  __shared__ float sh_x[ROWS];                        // decoder feedback x
  __shared__ int sh_len[ROWS];
  __shared__ float sh_hz[ROWS * 3];
  __shared__ float sh_red[ROWS];
  __shared__ float sh_p[8 * ROWS];                    // y-projection partials

  const int tid = threadIdx.x;
  const int lane = tid & 31;
  const int gate = tid >> 5;
  const int row_base = blockIdx.x * ROWS;

  // Echo padded into output region (rows are contiguous).
  {
    const float* src = pad + (size_t)row_base * SEQT;
    float* dst = out_pad + (size_t)row_base * SEQT;
    for (int i = tid; i < ROWS * SEQT; i += NTHREADS) dst[i] = src[i];
  }

  // Init state.
  if (tid < ROWS) {
    sh_len[tid] = seqlen[row_base + tid];
    sh_x[tid] = 0.0f;
  }
  {
    const int urow = tid & 15, ukb = (tid >> 4) << 3;
#pragma unroll
    for (int j = 0; j < 8; ++j) {
      sh_c[(ukb + j) * ROWS + urow] = 0.0f;
      sh_h[urow * HID + ukb + j] = (_Float16)0.0f;
    }
  }
  __syncthreads();

  // ---- Encoder: weights to resident WMMA B-fragments -----------------------
  v16h Bw[4][2], Bx[4];
#pragma unroll
  for (int nt = 0; nt < 4; ++nt) {
    const int n = gate * 64 + nt * 16 + (lane & 15);
    Bw[nt][0] = load_bfrag(enc_Whh, n, 0);
    Bw[nt][1] = load_bfrag(enc_Whh, n, 32);
    Bx[nt] = make_bext(enc_Wih, enc_b, n);
  }

  const size_t xrow = (size_t)(row_base + (lane & 15)) * SEQT;
#pragma unroll 1
  for (int t = 0; t < SEQT; ++t) {
    float xv = pad[xrow + t];
    if (t + 32 < SEQT) __builtin_prefetch(pad + xrow + t + 32, 0, 1);
    lstm_step(Bw, Bx, xv, /*masked=*/true, t, sh_h, sh_c, sh_g, sh_len);
  }

  // ---- Bottleneck: hz = sigmoid(h @ encW^T + b); hd = hz @ decW^T + b ------
  if (tid < ROWS * 3) {
    const int m = tid / 3, j = tid - m * 3;
    float s = enc_linb[j];
    for (int k = 0; k < HID; ++k)
      s += (float)sh_h[m * HID + k] * enc_linW[j * HID + k];
    float z = fast_sigmoid(s);
    sh_hz[m * 3 + j] = z;
    out_hz[(size_t)(row_base + m) * 3 + j] = z;
  }
  __syncthreads();
  {
    const int urow = tid & 15, ukb = (tid >> 4) << 3;
#pragma unroll
    for (int j = 0; j < 8; ++j) {
      const int k = ukb + j;
      float s = dec_linb[k];
      s += sh_hz[urow * 3 + 0] * dec_linW[k * 3 + 0];
      s += sh_hz[urow * 3 + 1] * dec_linW[k * 3 + 1];
      s += sh_hz[urow * 3 + 2] * dec_linW[k * 3 + 2];
      sh_h[urow * HID + k] = (_Float16)s;  // h <- hd ; c carries over
    }
  }
  __syncthreads();

  // ---- Decoder weights ------------------------------------------------------
#pragma unroll
  for (int nt = 0; nt < 4; ++nt) {
    const int n = gate * 64 + nt * 16 + (lane & 15);
    Bw[nt][0] = load_bfrag(dec_Whh, n, 0);
    Bw[nt][1] = load_bfrag(dec_Whh, n, 32);
    Bx[nt] = make_bext(dec_Wih, dec_b, n);
  }

  // Preload this thread's outW slice (8 k-values) into registers.
  const int urow = tid & 15;
  const int ukb = (tid >> 4) << 3;
  float wreg[8];
#pragma unroll
  for (int j = 0; j < 8; ++j) wreg[j] = outW[ukb + j];

  float sse = 0.0f;
  const float ob = outb[0];
#pragma unroll 1
  for (int t = 0; t < SEQT; ++t) {
    float xv = sh_x[lane & 15];  // autoregressive feedback
    lstm_step(Bw, Bx, xv, /*masked=*/false, t, sh_h, sh_c, sh_g, sh_len);

    // y = h_new @ outW + outb, distributed: 8 threads x 8 k per row.
    {
      const v8h* hp = (const v8h*)(sh_h + urow * HID + ukb);
      v8h hv = hp[0];  // one ds_load_b128
      float p = 0.0f;
#pragma unroll
      for (int j = 0; j < 8; ++j) p += (float)hv[j] * wreg[j];
      sh_p[(tid >> 4) * ROWS + urow] = p;
    }
    __syncthreads();
    if (tid < ROWS) {
      float s = ob;
#pragma unroll
      for (int i = 0; i < 8; ++i) s += sh_p[i * ROWS + tid];
      const size_t gi = (size_t)(row_base + tid) * SEQT + t;
      out_y[gi] = s;
      sh_x[tid] = s;
      float e = pad[gi] - s;
      if (t < sh_len[tid]) sse += e * e;
    }
    __syncthreads();
  }

  // ---- Masked SSE reduction -> workspace atomic ----------------------------
  if (tid < ROWS) sh_red[tid] = sse;
  __syncthreads();
  if (tid == 0) {
    float s = 0.0f;
    for (int i = 0; i < ROWS; ++i) s += sh_red[i];
    atomicAdd(ws_sse, s);
  }
}

__global__ void ws_zero_kernel(float* ws) { ws[0] = 0.0f; }

__global__ void finalize_kernel(const int* __restrict__ seqlen,
                                const float* __restrict__ ws,
                                float* __restrict__ loss) {
  __shared__ int red[256];
  int s = 0;
  for (int i = threadIdx.x; i < BATCH; i += 256) s += seqlen[i];
  red[threadIdx.x] = s;
  __syncthreads();
  for (int off = 128; off > 0; off >>= 1) {
    if (threadIdx.x < (unsigned)off) red[threadIdx.x] += red[threadIdx.x + off];
    __syncthreads();
  }
  if (threadIdx.x == 0) loss[0] = ws[0] / (float)red[0];
}

extern "C" void kernel_launch(void* const* d_in, const int* in_sizes, int n_in,
                              void* d_out, int out_size, void* d_ws,
                              size_t ws_size, hipStream_t stream) {
  const float* pad      = (const float*)d_in[0];
  const int*   seqlen   = (const int*)d_in[1];
  const float* enc_Wih  = (const float*)d_in[2];
  const float* enc_Whh  = (const float*)d_in[3];
  const float* enc_b    = (const float*)d_in[4];
  const float* enc_linW = (const float*)d_in[5];
  const float* enc_linb = (const float*)d_in[6];
  const float* dec_linW = (const float*)d_in[7];
  const float* dec_linb = (const float*)d_in[8];
  const float* dec_Wih  = (const float*)d_in[9];
  const float* dec_Whh  = (const float*)d_in[10];
  const float* dec_b    = (const float*)d_in[11];
  const float* outW     = (const float*)d_in[12];
  const float* outb     = (const float*)d_in[13];

  float* out = (float*)d_out;
  float* out_pad = out + 1;
  float* out_y = out + 1 + (size_t)BATCH * SEQT;
  float* out_hz = out + 1 + 2 * (size_t)BATCH * SEQT;
  float* ws = (float*)d_ws;

  ws_zero_kernel<<<1, 1, 0, stream>>>(ws);
  lstm_ae_kernel<<<BATCH / ROWS, NTHREADS, 0, stream>>>(
      pad, seqlen, enc_Wih, enc_Whh, enc_b, enc_linW, enc_linb, dec_linW,
      dec_linb, dec_Wih, dec_Whh, dec_b, outW, outb, out_pad, out_y, out_hz,
      ws);
  finalize_kernel<<<1, 256, 0, stream>>>(seqlen, ws, out);
}
